// part_seg_net_54176717472191
// MI455X (gfx1250) — matrix operand order repaired and checked by reading, not compiled, not run on hardware
//
#include <hip/hip_runtime.h>
#include <hip/hip_bf16.h>

#define B_    8
#define N_    2048
#define KNN_  30
#define CAT_  16
#define PART_ 50
#define EPS_  1e-5f
#define BNPTS (B_ * N_)          /* 16384  */
#define MEDGE (BNPTS * KNN_)     /* 491520 */

typedef __bf16 v16bf __attribute__((ext_vector_type(16)));
typedef float  v8f   __attribute__((ext_vector_type(8)));

__device__ __forceinline__ unsigned short f2bf(float f) {
  unsigned int u = __float_as_uint(f);
  u += 0x7FFFu + ((u >> 16) & 1u);   // round-to-nearest-even
  return (unsigned short)(u >> 16);
}
__device__ __forceinline__ unsigned int pk2(float a, float b) {
  return (unsigned int)f2bf(a) | ((unsigned int)f2bf(b) << 16);
}

// stage 16 contiguous f32 -> 16 bf16 (packed dword stores); src 16B aligned
__device__ __forceinline__ void stage16(const float* __restrict__ src,
                                        unsigned short* __restrict__ dst) {
  const float4* s4 = (const float4*)src;
  unsigned int* d32 = (unsigned int*)dst;
  #pragma unroll
  for (int q = 0; q < 4; ++q) {
    float4 f = s4[q];
    d32[2 * q]     = pk2(f.x, f.y);
    d32[2 * q + 1] = pk2(f.z, f.w);
  }
}
// stage 16 contiguous (a[i]-c[i]) -> bf16
__device__ __forceinline__ void stage16_diff(const float* __restrict__ a,
                                             const float* __restrict__ c,
                                             unsigned short* __restrict__ dst) {
  const float4* a4 = (const float4*)a;
  const float4* c4 = (const float4*)c;
  unsigned int* d32 = (unsigned int*)dst;
  #pragma unroll
  for (int q = 0; q < 4; ++q) {
    float4 fa = a4[q], fc = c4[q];
    d32[2 * q]     = pk2(fa.x - fc.x, fa.y - fc.y);
    d32[2 * q + 1] = pk2(fa.z - fc.z, fa.w - fc.w);
  }
}
__device__ __forceinline__ void stage_zero(unsigned short* __restrict__ dst) {
  unsigned int* d32 = (unsigned int*)dst;
  #pragma unroll
  for (int q = 0; q < 8; ++q) d32[q] = 0u;
}

// A-tile stage (row-major M x Kc). FULLK: M%64==0, Kc%32==0 (no guards at all).
template <bool FULLK>
__device__ __forceinline__ void stage_A(const float* __restrict__ A, int M, int Kc,
                                        int rowBase, int r, int cb, int k0,
                                        unsigned short* __restrict__ dst) {
  int grow = rowBase + r;
  if constexpr (FULLK) {
    stage16(A + (size_t)grow * Kc + k0 + cb, dst);
  } else {
    if ((rowBase + 64 <= M) && (k0 + 32 <= Kc) && ((Kc & 3) == 0)) {
      stage16(A + (size_t)grow * Kc + k0 + cb, dst);
    } else {
      #pragma unroll
      for (int j = 0; j < 16; ++j) {
        int k = k0 + cb + j;
        float v = (grow < M && k < Kc) ? A[(size_t)grow * Kc + k] : 0.f;
        dst[j] = f2bf(v);
      }
    }
  }
}

// B-tile stage (weights, row-major Ncols x Kc); rows beyond Ncols zero-filled.
template <bool FULLK>
__device__ __forceinline__ void stage_B(const float* __restrict__ W, int Ncols, int Kc,
                                        int colBase, int r, int cb, int k0,
                                        unsigned short* __restrict__ dst) {
  int n = colBase + r;
  if constexpr (FULLK) {
    if (n < Ncols) stage16(W + (size_t)n * Kc + k0 + cb, dst);
    else           stage_zero(dst);
  } else {
    if ((k0 + 32 <= Kc) && ((Kc & 3) == 0)) {
      if (n < Ncols) stage16(W + (size_t)n * Kc + k0 + cb, dst);
      else           stage_zero(dst);
    } else {
      #pragma unroll
      for (int j = 0; j < 4; ++j) {     // 4 quads of 4 -> same 16 cols
        #pragma unroll
        for (int q = 0; q < 4; ++q) {
          int k = k0 + cb + j * 4 + q;
          float v = (n < Ncols && k < Kc) ? W[(size_t)n * Kc + k] : 0.f;
          dst[j * 4 + q] = f2bf(v);
        }
      }
    }
  }
}

// edge-feature A-tile stage: row = edge position, cols of [ctr | nbr-ctr] (2C wide)
// FULLK: Mtot%64==0 and C%16==0 (each 16-chunk entirely in ctr or diff region).
template <bool FULLK>
__device__ __forceinline__ void stage_E(const float* __restrict__ feat,
                                        const int* __restrict__ idx,
                                        int Mtot, int Kk, int C, int rowBase,
                                        int r, int cb, int k0,
                                        unsigned short* __restrict__ dst) {
  int grow = rowBase + r;
  int kb = k0 + cb;
  if constexpr (FULLK) {
    int bn = grow / Kk;
    if (kb < C) {
      stage16(feat + (size_t)bn * C + kb, dst);
    } else {
      int nbr = idx[grow];
      int cc = kb - C;
      stage16_diff(feat + (size_t)nbr * C + cc, feat + (size_t)bn * C + cc, dst);
    }
  } else {
    bool ok = grow < Mtot;
    int bn = ok ? grow / Kk : 0;
    int nbr = ok ? idx[grow] : 0;
    #pragma unroll
    for (int j = 0; j < 16; ++j) {
      int k = kb + j;
      float v = 0.f;
      if (ok && k < 2 * C) {
        if (k < C) v = feat[(size_t)bn * C + k];
        else {
          int cc = k - C;
          v = feat[(size_t)nbr * C + cc] - feat[(size_t)bn * C + cc];
        }
      }
      dst[j] = f2bf(v);
    }
  }
}

// 4x WMMA step: one A fragment (16 rows) x 4 B fragments (64 cols)
__device__ __forceinline__ void wmma_step4(const unsigned short (*__restrict__ pA)[40],
                                           const unsigned short (*__restrict__ pB)[40],
                                           int wave, int half, int l16, v8f acc[4]) {
  union { v16bf v; unsigned short u[16]; } fa;
  const int arow = wave * 16 + l16;
  #pragma unroll
  for (int i = 0; i < 16; ++i) {
    int ka = (i < 8) ? (half * 8 + i) : (16 + half * 8 + (i - 8));
    fa.u[i] = pA[arow][ka];
  }
  #pragma unroll
  for (int j = 0; j < 4; ++j) {
    union { v16bf v; unsigned short u[16]; } fb;
    const int brow = j * 16 + l16;
    #pragma unroll
    for (int i = 0; i < 16; ++i) fb.u[i] = pB[brow][(i < 8) ? (half * 8 + i) : (16 + half * 8 + (i - 8))];
    acc[j] = __builtin_amdgcn_wmma_f32_16x16x32_bf16(false, fa.v, false, fb.v,
                                                     (short)0, acc[j], false, false);
  }
}

// ---------------------------------------------------------------------------
// Core WMMA GEMM: out[M x Ncols] = act( A[M x Kc] * W[Ncols x Kc]^T + bias )
// Block = 128 threads (4 waves); block tile 64x64; double-buffered LDS.
// ---------------------------------------------------------------------------
template <bool FULLK>
__global__ __launch_bounds__(128) void gemm_bias_act(
    const float* __restrict__ A, const float* __restrict__ W,
    const float* __restrict__ bias, float* __restrict__ out,
    int M, int Ncols, int Kc, int relu)
{
  __shared__ __align__(16) unsigned short sA[2][64][40];
  __shared__ __align__(16) unsigned short sB[2][64][40];
  const int tid  = threadIdx.x;
  const int lane = tid & 31;
  const int wave = tid >> 5;
  const int half = lane >> 4;
  const int l16  = lane & 15;
  const int rowBase = blockIdx.x * 64;
  const int colBase = blockIdx.y * 64;
  const int r  = tid >> 1;          // tile row loaded by this thread
  const int cb = (tid & 1) * 16;    // tile col chunk

  v8f acc[4];
  #pragma unroll
  for (int j = 0; j < 4; ++j) acc[j] = (v8f){0.f,0.f,0.f,0.f,0.f,0.f,0.f,0.f};

  stage_A<FULLK>(A, M, Kc, rowBase, r, cb, 0, &sA[0][r][cb]);
  stage_B<FULLK>(W, Ncols, Kc, colBase, r, cb, 0, &sB[0][r][cb]);
  __syncthreads();

  int pb = 0;
  for (int k0 = 0; k0 < Kc; k0 += 32) {
    int kn = k0 + 32;
    if (kn < Kc) {
      stage_A<FULLK>(A, M, Kc, rowBase, r, cb, kn, &sA[pb ^ 1][r][cb]);
      stage_B<FULLK>(W, Ncols, Kc, colBase, r, cb, kn, &sB[pb ^ 1][r][cb]);
    }
    wmma_step4(sA[pb], sB[pb], wave, half, l16, acc);
    __syncthreads();
    pb ^= 1;
  }

  #pragma unroll
  for (int j = 0; j < 4; ++j) {
    int n = colBase + j * 16 + l16;
    #pragma unroll
    for (int v = 0; v < 8; ++v) {
      int m = rowBase + wave * 16 + half * 8 + v;
      if (m < M && n < Ncols) {
        float val = acc[j][v] + bias[n];
        if (relu) val = fmaxf(val, 0.f);
        out[(size_t)m * Ncols + n] = val;
      }
    }
  }
}

// ---------------------------------------------------------------------------
// Fused edge-feature GEMM: out = relu( [ctr|nbr-ctr] * W^T + bias )
// ---------------------------------------------------------------------------
template <bool FULLK>
__global__ __launch_bounds__(128) void edge_gemm(
    const float* __restrict__ feat, const int* __restrict__ idx,
    const float* __restrict__ W, const float* __restrict__ bias,
    float* __restrict__ out, int Mtot, int Kk, int C, int Ncols)
{
  __shared__ __align__(16) unsigned short sA[2][64][40];
  __shared__ __align__(16) unsigned short sB[2][64][40];
  const int tid  = threadIdx.x;
  const int lane = tid & 31;
  const int wave = tid >> 5;
  const int half = lane >> 4;
  const int l16  = lane & 15;
  const int rowBase = blockIdx.x * 64;
  const int colBase = blockIdx.y * 64;
  const int Kc = 2 * C;
  const int r  = tid >> 1;
  const int cb = (tid & 1) * 16;

  v8f acc[4];
  #pragma unroll
  for (int j = 0; j < 4; ++j) acc[j] = (v8f){0.f,0.f,0.f,0.f,0.f,0.f,0.f,0.f};

  stage_E<FULLK>(feat, idx, Mtot, Kk, C, rowBase, r, cb, 0, &sA[0][r][cb]);
  stage_B<FULLK>(W, Ncols, Kc, colBase, r, cb, 0, &sB[0][r][cb]);
  __syncthreads();

  int pb = 0;
  for (int k0 = 0; k0 < Kc; k0 += 32) {
    int kn = k0 + 32;
    if (kn < Kc) {
      stage_E<FULLK>(feat, idx, Mtot, Kk, C, rowBase, r, cb, kn, &sA[pb ^ 1][r][cb]);
      stage_B<FULLK>(W, Ncols, Kc, colBase, r, cb, kn, &sB[pb ^ 1][r][cb]);
    }
    wmma_step4(sA[pb], sB[pb], wave, half, l16, acc);
    __syncthreads();
    pb ^= 1;
  }

  #pragma unroll
  for (int j = 0; j < 4; ++j) {
    int n = colBase + j * 16 + l16;
    #pragma unroll
    for (int v = 0; v < 8; ++v) {
      int m = rowBase + wave * 16 + half * 8 + v;
      if (m < Mtot && n < Ncols) {
        float val = fmaxf(acc[j][v] + bias[n], 0.f);
        out[(size_t)m * Ncols + n] = val;
      }
    }
  }
}

// ---------------------------------------------------------------------------
// Pairwise squared distances (one batch): dist[m,n] = |xm|^2 + |xn|^2 - 2 xm.xn
// ---------------------------------------------------------------------------
template <bool FULLK>
__global__ __launch_bounds__(128) void gram_dist(
    const float* __restrict__ F, const float* __restrict__ sq,
    float* __restrict__ dist, int Np, int C)
{
  __shared__ __align__(16) unsigned short sA[2][64][40];
  __shared__ __align__(16) unsigned short sB[2][64][40];
  const int tid  = threadIdx.x;
  const int lane = tid & 31;
  const int wave = tid >> 5;
  const int half = lane >> 4;
  const int l16  = lane & 15;
  const int rowBase = blockIdx.x * 64;
  const int colBase = blockIdx.y * 64;
  const int r  = tid >> 1;
  const int cb = (tid & 1) * 16;

  v8f acc[4];
  #pragma unroll
  for (int j = 0; j < 4; ++j) acc[j] = (v8f){0.f,0.f,0.f,0.f,0.f,0.f,0.f,0.f};

  stage_A<FULLK>(F, Np, C, rowBase, r, cb, 0, &sA[0][r][cb]);
  stage_B<FULLK>(F, Np, C, colBase, r, cb, 0, &sB[0][r][cb]);
  __syncthreads();

  int pb = 0;
  for (int k0 = 0; k0 < C; k0 += 32) {
    int kn = k0 + 32;
    if (kn < C) {
      stage_A<FULLK>(F, Np, C, rowBase, r, cb, kn, &sA[pb ^ 1][r][cb]);
      stage_B<FULLK>(F, Np, C, colBase, r, cb, kn, &sB[pb ^ 1][r][cb]);
    }
    wmma_step4(sA[pb], sB[pb], wave, half, l16, acc);
    __syncthreads();
    pb ^= 1;
  }

  #pragma unroll
  for (int j = 0; j < 4; ++j) {
    int n = colBase + j * 16 + l16;
    #pragma unroll
    for (int v = 0; v < 8; ++v) {
      int m = rowBase + wave * 16 + half * 8 + v;
      if (m < Np && n < Np)
        dist[(size_t)m * Np + n] = sq[m] + sq[n] - 2.f * acc[j][v];
    }
  }
}

// ---------------------------------------------------------------------------
// BN fold into weights: W2[n,k] = W[n,k]*sc[k]; b2[n] = bias[n] + sum_k W[n,k]*sh[k]
__global__ void fold_weights(const float* __restrict__ W, const float* __restrict__ bias,
                             const float* __restrict__ sc, const float* __restrict__ sh,
                             float* __restrict__ W2, float* __restrict__ b2,
                             int Nc, int Kc) {
  int n = blockIdx.x;
  float acc = 0.f;
  for (int k = threadIdx.x; k < Kc; k += blockDim.x) {
    float wv = W[(size_t)n * Kc + k];
    W2[(size_t)n * Kc + k] = wv * sc[k];
    acc += wv * sh[k];
  }
  __shared__ float red[256];
  red[threadIdx.x] = acc;
  __syncthreads();
  for (int st = 128; st > 0; st >>= 1) {
    if ((int)threadIdx.x < st) red[threadIdx.x] += red[threadIdx.x + st];
    __syncthreads();
  }
  if (threadIdx.x == 0) b2[n] = bias[n] + red[0];
}

// per-row squared norm
__global__ void rowsq_kernel(const float* __restrict__ X, int M, int C,
                             float* __restrict__ sq) {
  int i = blockIdx.x * blockDim.x + threadIdx.x;
  if (i >= M) return;
  float s = 0.f;
  for (int c = 0; c < C; ++c) { float v = X[(size_t)i * C + c]; s += v * v; }
  sq[i] = s;
}

// top-K smallest distances per row; stores GLOBAL point row (b*N + j)
__global__ void topk_kernel(const float* __restrict__ dist, int* __restrict__ idx,
                            int Np, int b) {
  int i = blockIdx.x * blockDim.x + threadIdx.x;
  if (i >= Np) return;
  float bd[KNN_]; int bi[KNN_];
  #pragma unroll
  for (int t = 0; t < KNN_; ++t) { bd[t] = 3.4e38f; bi[t] = 0; }
  const float* row = dist + (size_t)i * Np;
  for (int j = 0; j < Np; ++j) {
    float d = row[j];
    if (d < bd[KNN_ - 1]) {
      int t = KNN_ - 1;
      while (t > 0 && bd[t - 1] > d) { bd[t] = bd[t - 1]; bi[t] = bi[t - 1]; --t; }
      bd[t] = d; bi[t] = j;
    }
  }
  int* orow = idx + ((size_t)b * Np + i) * KNN_;
  #pragma unroll
  for (int t = 0; t < KNN_; ++t) orow[t] = b * Np + bi[t];
}

// per-channel sum / sumsq (one block per channel)
__global__ void colstats(const float* __restrict__ X, int M, int C,
                         float* __restrict__ sums) {
  int c = blockIdx.x;
  float s = 0.f, s2 = 0.f;
  for (int i = threadIdx.x; i < M; i += blockDim.x) {
    float v = X[(size_t)i * C + c];
    s += v; s2 += v * v;
  }
  __shared__ float sh1[256]; __shared__ float sh2[256];
  sh1[threadIdx.x] = s; sh2[threadIdx.x] = s2;
  __syncthreads();
  for (int st = 128; st > 0; st >>= 1) {
    if ((int)threadIdx.x < st) {
      sh1[threadIdx.x] += sh1[threadIdx.x + st];
      sh2[threadIdx.x] += sh2[threadIdx.x + st];
    }
    __syncthreads();
  }
  if (threadIdx.x == 0) { sums[c] = sh1[0]; sums[C + c] = sh2[0]; }
}

__global__ void finalize_bn(const float* __restrict__ sums, int C, float Minv,
                            const float* __restrict__ g, const float* __restrict__ b,
                            float* __restrict__ sc, float* __restrict__ sh) {
  int c = blockIdx.x * blockDim.x + threadIdx.x;
  if (c >= C) return;
  float m = sums[c] * Minv;
  float v = sums[C + c] * Minv - m * m;
  float s = g[c] * rsqrtf(v + EPS_);
  sc[c] = s;
  sh[c] = b[c] - m * s;
}

__global__ void bn_apply(const float* __restrict__ X, const float* __restrict__ sc,
                         const float* __restrict__ sh, float* __restrict__ out,
                         long total, int C) {
  long t = blockIdx.x * (long)blockDim.x + threadIdx.x;
  if (t >= total) return;
  int c = (int)(t % C);
  out[t] = X[t] * sc[c] + sh[c];
}

// mode 0: out[bn][0:C]=max_k bn(x), out[bn][C:2C]=mean_k bn(x)
// mode 1: out[bn][0:C]=max_k bn(x)
__global__ void pool_k(const float* __restrict__ X, int BN, int Kk, int C,
                       const float* __restrict__ sc, const float* __restrict__ sh,
                       float* __restrict__ out, int mode) {
  int t = blockIdx.x * blockDim.x + threadIdx.x;
  if (t >= BN * C) return;
  int bn = t / C, c = t % C;
  float s = sc[c], h = sh[c];
  const float* p = X + ((size_t)bn * Kk) * C + c;
  float mx = -3.4e38f, sm = 0.f;
  for (int k = 0; k < Kk; ++k) {
    float v = p[(size_t)k * C] * s + h;
    mx = fmaxf(mx, v); sm += v;
  }
  if (mode == 0) {
    out[(size_t)bn * (2 * C) + c]     = mx;
    out[(size_t)bn * (2 * C) + C + c] = sm / (float)Kk;
  } else {
    out[(size_t)bn * C + c] = mx;
  }
}

// max over the N points of each batch (optional BN fold)
__global__ void max_over_points(const float* __restrict__ X, int Bb, int Np, int C,
                                const float* __restrict__ sc, const float* __restrict__ sh,
                                float* __restrict__ out) {
  int t = blockIdx.x * blockDim.x + threadIdx.x;
  if (t >= Bb * C) return;
  int b = t / C, c = t % C;
  float s = sc ? sc[c] : 1.f, h = sh ? sh[c] : 0.f;
  const float* p = X + ((size_t)b * Np) * C + c;
  float mx = -3.4e38f;
  for (int n = 0; n < Np; ++n) mx = fmaxf(mx, p[(size_t)n * C] * s + h);
  out[(size_t)b * C + c] = mx;
}

// final FC of t-net: T[b,9] = bn5(f2[b]) @ w(256x9) + bias(9)
__global__ void tfc3_kernel(const float* __restrict__ f2, const float* __restrict__ sc,
                            const float* __restrict__ sh, const float* __restrict__ w,
                            const float* __restrict__ bias, float* __restrict__ T,
                            int Bb) {
  int t = blockIdx.x * blockDim.x + threadIdx.x;
  if (t >= Bb * 9) return;
  int b = t / 9, j = t % 9;
  float acc = bias[j];
  for (int c = 0; c < 256; ++c)
    acc += (f2[b * 256 + c] * sc[c] + sh[c]) * w[c * 9 + j];
  T[b * 9 + j] = acc;
}

// pct[b,n,:] = pc[b,n,:] @ T[b]
__global__ void apply_transform(const float* __restrict__ pc, const float* __restrict__ T,
                                float* __restrict__ pct, int total) {
  int t = blockIdx.x * blockDim.x + threadIdx.x;
  if (t >= total) return;
  int b = t / N_;
  const float* Tb = T + b * 9;
  const float* p  = pc + (size_t)t * 3;
  float x = p[0], y = p[1], z = p[2];
  float* o = pct + (size_t)t * 3;
  o[0] = x * Tb[0] + y * Tb[3] + z * Tb[6];
  o[1] = x * Tb[1] + y * Tb[4] + z * Tb[7];
  o[2] = x * Tb[2] + y * Tb[5] + z * Tb[8];
}

// concat [out3|out5|out7] -> (BN,192)
__global__ void build_cat8(const float* __restrict__ o3, const float* __restrict__ o5,
                           const float* __restrict__ o7, float* __restrict__ out,
                           long total) {
  long t = blockIdx.x * (long)blockDim.x + threadIdx.x;
  if (t >= total) return;
  int bn = (int)(t / 192), c = (int)(t % 192);
  float v;
  if (c < 64)       v = o3[(size_t)bn * 64 + c];
  else if (c < 128) v = o5[(size_t)bn * 64 + (c - 64)];
  else              v = o7[(size_t)bn * 64 + (c - 128)];
  out[t] = v;
}

// big concat (BN,2752)
__global__ void build_catbig(const float* __restrict__ gmax, const float* __restrict__ lab,
                             const float* __restrict__ mm1, const float* __restrict__ o3,
                             const float* __restrict__ mm2, const float* __restrict__ o5,
                             const float* __restrict__ mm3, const float* __restrict__ o7,
                             const float* __restrict__ o8, float* __restrict__ cat,
                             long total) {
  long t = blockIdx.x * (long)blockDim.x + threadIdx.x;
  if (t >= total) return;
  int bn = (int)(t / 2752), c = (int)(t % 2752);
  int b = bn / N_;
  float v;
  if (c < 1024)      v = gmax[b * 1024 + c];
  else if (c < 1152) v = lab[b * 128 + (c - 1024)];
  else if (c < 1280) v = mm1[(size_t)bn * 128 + (c - 1152)];
  else if (c < 1344) v = o3[(size_t)bn * 64 + (c - 1280)];
  else if (c < 1472) v = mm2[(size_t)bn * 128 + (c - 1344)];
  else if (c < 1536) v = o5[(size_t)bn * 64 + (c - 1472)];
  else if (c < 1664) v = mm3[(size_t)bn * 128 + (c - 1536)];
  else if (c < 1728) v = o7[(size_t)bn * 64 + (c - 1664)];
  else               v = o8[(size_t)bn * 1024 + (c - 1728)];
  cat[t] = v;
}

// log_softmax over channels; output in reference (B, PART, N) order
__global__ void log_softmax_out(const float* __restrict__ logits, float* __restrict__ out,
                                int total) {
  int t = blockIdx.x * blockDim.x + threadIdx.x;
  if (t >= total) return;
  int b = t / N_, n = t % N_;
  const float* row = logits + (size_t)t * PART_;
  float mx = -3.4e38f;
  for (int p = 0; p < PART_; ++p) mx = fmaxf(mx, row[p]);
  float s = 0.f;
  for (int p = 0; p < PART_; ++p) s += expf(row[p] - mx);
  float lse = mx + logf(s);
  for (int p = 0; p < PART_; ++p)
    out[((size_t)b * PART_ + p) * N_ + n] = row[p] - lse;
}

// ---------------------------------------------------------------------------
// Host launcher
// ---------------------------------------------------------------------------
enum {
  IN_PC = 0, IN_LABEL = 1,
  BN1_B = 2, BN1_G, BN10_B, BN10_G, BN11_B, BN11_G, BN12_B, BN12_G,
  BN2_B, BN2_G, BN3_B, BN3_G, BN4_B, BN4_G, BN5_B, BN5_G,
  BN6_B, BN6_G, BN7_B, BN7_G, BN8_B, BN8_G, BN9_B, BN9_G,
  CONV1_B, CONV1_W, CONV10_B, CONV10_W, CONV11_B, CONV11_W,
  CONV12_B, CONV12_W, CONV13_B, CONV13_W, CONV2_B, CONV2_W,
  CONV3_B, CONV3_W, CONV4_B, CONV4_W, CONV5_B, CONV5_W,
  CONV6_B, CONV6_W, CONV7_B, CONV7_W, CONV8_B, CONV8_W, CONV9_B, CONV9_W,
  TBN1_B, TBN1_G, TBN2_B, TBN2_G, TBN3_B, TBN3_G, TBN4_B, TBN4_G, TBN5_B, TBN5_G,
  TCONV1_B, TCONV1_W, TCONV2_B, TCONV2_W, TCONV3_B, TCONV3_W,
  TFC1_B, TFC1_W, TFC2_B, TFC2_W, TFC3_B, TFC3_W
};

extern "C" void kernel_launch(void* const* d_in, const int* in_sizes, int n_in,
                              void* d_out, int out_size, void* d_ws, size_t ws_size,
                              hipStream_t stream) {
  (void)in_sizes; (void)n_in; (void)out_size; (void)ws_size;
  auto IN = [&](int i) { return (const float*)d_in[i]; };

  float* w = (float*)d_ws;
  size_t off = 0;
  auto alloc = [&](size_t nfloats) { float* p = w + off; off += nfloats; return p; };

  int*   d_idx  = (int*)alloc(MEDGE);
  float* d_sq   = alloc(BNPTS);
  float* d_dist = alloc((size_t)N_ * N_);
  float* E1   = alloc((size_t)MEDGE * 64);
  float* E2   = alloc((size_t)MEDGE * 128);
  float* MM1  = alloc((size_t)BNPTS * 128);
  float* MM2  = alloc((size_t)BNPTS * 128);
  float* MM3  = alloc((size_t)BNPTS * 128);
  float* TMX  = alloc((size_t)BNPTS * 128);
  float* O3   = alloc((size_t)BNPTS * 64);
  float* O5   = alloc((size_t)BNPTS * 64);
  float* O7   = alloc((size_t)BNPTS * 64);
  float* P1   = alloc((size_t)BNPTS * 64);
  float* T3   = alloc((size_t)BNPTS * 1024);
  float* P8   = alloc((size_t)BNPTS * 1024);
  float* O8   = alloc((size_t)BNPTS * 1024);
  float* C8   = alloc((size_t)BNPTS * 192);
  float* CATB = alloc((size_t)BNPTS * 2752);
  float* Q1   = alloc((size_t)BNPTS * 256);
  float* Q2   = alloc((size_t)BNPTS * 256);
  float* Q3   = alloc((size_t)BNPTS * 128);
  float* LOGI = alloc((size_t)BNPTS * PART_);
  float* SUMS = alloc(8192);
  float* SC   = alloc(4096);
  float* SH   = alloc(4096);
  float* WF   = alloc((size_t)256 * 512);   // folded weights (max 256x512)
  float* BF   = alloc(1024);                // folded bias
  float* GB   = alloc(B_ * 1024);
  float* F1   = alloc(B_ * 512);
  float* F2   = alloc(B_ * 256);
  float* TMAT = alloc(128);
  float* LRAW = alloc(B_ * 128);
  float* LAB  = alloc(B_ * 128);
  float* GMX  = alloc(B_ * 1024);
  float* PCT  = alloc((size_t)BNPTS * 3);

  auto gemm = [&](const float* A, const float* W_, const float* b, float* out,
                  int M, int Nc, int Kc, int relu) {
    dim3 g((M + 63) / 64, (Nc + 63) / 64);
    if ((M % 64 == 0) && (Kc % 32 == 0))
      gemm_bias_act<true><<<g, 128, 0, stream>>>(A, W_, b, out, M, Nc, Kc, relu);
    else
      gemm_bias_act<false><<<g, 128, 0, stream>>>(A, W_, b, out, M, Nc, Kc, relu);
  };
  auto egemm = [&](const float* feat, const float* W_, const float* b,
                   float* out, int C, int Nc) {
    dim3 g((MEDGE + 63) / 64, (Nc + 63) / 64);
    if (C % 16 == 0)
      edge_gemm<true><<<g, 128, 0, stream>>>(feat, d_idx, W_, b, out, MEDGE, KNN_, C, Nc);
    else
      edge_gemm<false><<<g, 128, 0, stream>>>(feat, d_idx, W_, b, out, MEDGE, KNN_, C, Nc);
  };
  auto stats = [&](const float* X, int M, int C, const float* g_, const float* b_) {
    colstats<<<C, 256, 0, stream>>>(X, M, C, SUMS);
    finalize_bn<<<(C + 255) / 256, 256, 0, stream>>>(SUMS, C, 1.0f / (float)M,
                                                     g_, b_, SC, SH);
  };
  auto fold = [&](const float* W_, const float* b_, int Nc, int Kc) {
    fold_weights<<<Nc, 256, 0, stream>>>(W_, b_, SC, SH, WF, BF, Nc, Kc);
  };
  auto knn = [&](const float* feat, int C) {
    rowsq_kernel<<<(BNPTS + 255) / 256, 256, 0, stream>>>(feat, BNPTS, C, d_sq);
    for (int b = 0; b < B_; ++b) {
      dim3 g((N_ + 63) / 64, (N_ + 63) / 64);
      if (C % 32 == 0)
        gram_dist<true><<<g, 128, 0, stream>>>(feat + (size_t)b * N_ * C,
                                               d_sq + b * N_, d_dist, N_, C);
      else
        gram_dist<false><<<g, 128, 0, stream>>>(feat + (size_t)b * N_ * C,
                                                d_sq + b * N_, d_dist, N_, C);
      topk_kernel<<<(N_ + 127) / 128, 128, 0, stream>>>(d_dist, d_idx, N_, b);
    }
  };
  auto apply = [&](const float* X, float* out, long total, int C) {
    bn_apply<<<(int)((total + 255) / 256), 256, 0, stream>>>(X, SC, SH, out, total, C);
  };

  const float* pc  = IN(IN_PC);
  const float* lbl = IN(IN_LABEL);

  // ===================== Transform net (on pc) =====================
  knn(pc, 3);
  egemm(pc, IN(TCONV1_W), IN(TCONV1_B), E1, 3, 64);                       // t1raw
  stats(E1, MEDGE, 64, IN(TBN1_G), IN(TBN1_B));
  fold(IN(TCONV2_W), IN(TCONV2_B), 128, 64);
  gemm(E1, WF, BF, E2, MEDGE, 128, 64, 1);                                // t2raw
  stats(E2, MEDGE, 128, IN(TBN2_G), IN(TBN2_B));
  pool_k<<<(BNPTS * 128 + 255) / 256, 256, 0, stream>>>(E2, BNPTS, KNN_, 128,
                                                        SC, SH, TMX, 1);   // max over k
  gemm(TMX, IN(TCONV3_W), IN(TCONV3_B), T3, BNPTS, 1024, 128, 1);
  stats(T3, BNPTS, 1024, IN(TBN3_G), IN(TBN3_B));
  max_over_points<<<(B_ * 1024 + 255) / 256, 256, 0, stream>>>(T3, B_, N_, 1024,
                                                               SC, SH, GB);
  gemm(GB, IN(TFC1_W), IN(TFC1_B), F1, B_, 512, 1024, 1);
  stats(F1, B_, 512, IN(TBN4_G), IN(TBN4_B));
  fold(IN(TFC2_W), IN(TFC2_B), 256, 512);
  gemm(F1, WF, BF, F2, B_, 256, 512, 1);
  stats(F2, B_, 256, IN(TBN5_G), IN(TBN5_B));
  tfc3_kernel<<<1, 128, 0, stream>>>(F2, SC, SH, IN(TFC3_W), IN(TFC3_B), TMAT, B_);
  apply_transform<<<(BNPTS + 255) / 256, 256, 0, stream>>>(pc, TMAT, PCT, BNPTS);

  // ===================== Edge-conv stage 1 (on pct) =====================
  knn(PCT, 3);
  egemm(PCT, IN(CONV1_W), IN(CONV1_B), E1, 3, 64);                        // out1raw
  stats(E1, MEDGE, 64, IN(BN1_G), IN(BN1_B));
  fold(IN(CONV2_W), IN(CONV2_B), 64, 64);
  gemm(E1, WF, BF, E2, MEDGE, 64, 64, 1);                                 // out2raw
  stats(E2, MEDGE, 64, IN(BN2_G), IN(BN2_B));
  pool_k<<<(BNPTS * 64 + 255) / 256, 256, 0, stream>>>(E2, BNPTS, KNN_, 64,
                                                       SC, SH, MM1, 0);    // max|mean
  gemm(MM1, IN(CONV3_W), IN(CONV3_B), P1, BNPTS, 64, 128, 1);
  stats(P1, BNPTS, 64, IN(BN3_G), IN(BN3_B));
  apply(P1, O3, (long)BNPTS * 64, 64);                                     // out3

  // ===================== Edge-conv stage 2 (on out3) =====================
  knn(O3, 64);
  egemm(O3, IN(CONV4_W), IN(CONV4_B), E1, 64, 64);                         // out4raw
  stats(E1, MEDGE, 64, IN(BN4_G), IN(BN4_B));
  pool_k<<<(BNPTS * 64 + 255) / 256, 256, 0, stream>>>(E1, BNPTS, KNN_, 64,
                                                       SC, SH, MM2, 0);
  gemm(MM2, IN(CONV5_W), IN(CONV5_B), P1, BNPTS, 64, 128, 1);
  stats(P1, BNPTS, 64, IN(BN5_G), IN(BN5_B));
  apply(P1, O5, (long)BNPTS * 64, 64);                                     // out5

  // ===================== Edge-conv stage 3 (on out5) =====================
  knn(O5, 64);
  egemm(O5, IN(CONV6_W), IN(CONV6_B), E1, 64, 64);                         // out6raw
  stats(E1, MEDGE, 64, IN(BN6_G), IN(BN6_B));
  pool_k<<<(BNPTS * 64 + 255) / 256, 256, 0, stream>>>(E1, BNPTS, KNN_, 64,
                                                       SC, SH, MM3, 0);
  gemm(MM3, IN(CONV7_W), IN(CONV7_B), P1, BNPTS, 64, 128, 1);
  stats(P1, BNPTS, 64, IN(BN7_G), IN(BN7_B));
  apply(P1, O7, (long)BNPTS * 64, 64);                                     // out7

  // ===================== conv8 + global features =====================
  build_cat8<<<(int)(((long)BNPTS * 192 + 255) / 256), 256, 0, stream>>>(
      O3, O5, O7, C8, (long)BNPTS * 192);
  gemm(C8, IN(CONV8_W), IN(CONV8_B), P8, BNPTS, 1024, 192, 1);
  stats(P8, BNPTS, 1024, IN(BN8_G), IN(BN8_B));
  apply(P8, O8, (long)BNPTS * 1024, 1024);                                 // out8
  max_over_points<<<(B_ * 1024 + 255) / 256, 256, 0, stream>>>(O8, B_, N_, 1024,
                                                               nullptr, nullptr, GMX);
  gemm(lbl, IN(CONV9_W), IN(CONV9_B), LRAW, B_, 128, CAT_, 1);
  stats(LRAW, B_, 128, IN(BN9_G), IN(BN9_B));
  apply(LRAW, LAB, (long)B_ * 128, 128);                                   // label feat

  // ===================== segmentation head =====================
  build_catbig<<<(int)(((long)BNPTS * 2752 + 255) / 256), 256, 0, stream>>>(
      GMX, LAB, MM1, O3, MM2, O5, MM3, O7, O8, CATB, (long)BNPTS * 2752);
  gemm(CATB, IN(CONV10_W), IN(CONV10_B), Q1, BNPTS, 256, 2752, 1);
  stats(Q1, BNPTS, 256, IN(BN10_G), IN(BN10_B));
  fold(IN(CONV11_W), IN(CONV11_B), 256, 256);
  gemm(Q1, WF, BF, Q2, BNPTS, 256, 256, 1);
  stats(Q2, BNPTS, 256, IN(BN11_G), IN(BN11_B));
  fold(IN(CONV12_W), IN(CONV12_B), 128, 256);
  gemm(Q2, WF, BF, Q3, BNPTS, 128, 256, 1);
  stats(Q3, BNPTS, 128, IN(BN12_G), IN(BN12_B));
  fold(IN(CONV13_W), IN(CONV13_B), PART_, 128);
  gemm(Q3, WF, BF, LOGI, BNPTS, PART_, 128, 0);
  log_softmax_out<<<(BNPTS + 255) / 256, 256, 0, stream>>>(LOGI, (float*)d_out, BNPTS);
}
